// AttLayer_88880053224157
// MI455X (gfx1250) — matrix operand order repaired and checked by reading, not compiled, not run on hardware
//
#include <hip/hip_runtime.h>
#include <math.h>

typedef __bf16 bf16;
typedef __attribute__((ext_vector_type(16))) __bf16 v16bf;
typedef __attribute__((ext_vector_type(8)))  __bf16 v8bf;
typedef __attribute__((ext_vector_type(8)))  float  v8f;
typedef __attribute__((ext_vector_type(4)))  unsigned int u32x4;
typedef __attribute__((ext_vector_type(8)))  int i32x8;
typedef __attribute__((ext_vector_type(4)))  int i32x4;

#define L_TOT 65536
#define QD    64
#define CH    32
#define BLK   512
#define HALF  256
#define WW    1024   // window length
#define LPAD  (L_TOT + 2*HALF)
#define KSTR  40     // k_lds row stride (halves): 80B = 16 data dwords + 4 pad dwords
#define VSTR  1032   // v_lds row stride (halves): 2064B = 16*129, odd dword-phase
#define SSTR  40     // per-wave scratch row stride (halves)
#define LOG_EPS (-20.72326583694641f)   // logf(1e-9)

__device__ inline v8f wmma_bf16(v16bf a, v16bf b, v8f c) {
    return __builtin_amdgcn_wmma_f32_16x16x32_bf16(false, a, false, b,
                                                   (short)0, c, false, false);
}

__device__ inline v16bf ld16(const bf16* p0, const bf16* p1) {
    v8bf a = *(const v8bf*)p0;
    v8bf b = *(const v8bf*)p1;
    v16bf r;
#pragma unroll
    for (int i = 0; i < 8; ++i) { r[i] = a[i]; r[i + 8] = b[i]; }
    return r;
}

__device__ inline float winmask(const float* __restrict__ mask, int p) {
    int l = p - HALF;
    return (l >= 0 && l < L_TOT) ? mask[l] : 0.f;
}

// ---------------------------------------------------------------- pad halos
__global__ __launch_bounds__(256) void pad_kernel(bf16* kt, bf16* vt) {
    int idx = blockIdx.x * 256 + threadIdx.x;     // 2 * 512 * 32 = 32768
    if (idx >= 2 * 512 * CH) return;
    int which = idx / (512 * CH);
    int rem   = idx % (512 * CH);
    int r512  = rem / CH;
    int ch    = rem % CH;
    size_t row = (r512 < HALF) ? (size_t)r512 : (size_t)(r512 + L_TOT);
    bf16* p = which ? vt : kt;
    p[row * CH + ch] = (bf16)0.f;
}

// ------------------------------------------------------- q/k/v projections
__global__ __launch_bounds__(256) void proj_kernel(
    const float* __restrict__ x1,
    const float* __restrict__ Wq, const float* __restrict__ bq,
    const float* __restrict__ Wk, const float* __restrict__ bk,
    const float* __restrict__ Wv, const float* __restrict__ bv,
    bf16* __restrict__ qt, bf16* __restrict__ kt, bf16* __restrict__ vt)
{
    __shared__ float wsh[3 * CH * QD];   // 24 KB
    __shared__ float bsh[3 * CH];
    __shared__ float xs[QD * 256];       // 64 KB

    const int tid  = threadIdx.x;
    const int pos0 = blockIdx.x * 256;

    for (int i = tid; i < 3 * CH * QD; i += 256) {
        const float* src = (i < CH * QD) ? Wq : (i < 2 * CH * QD ? Wk : Wv);
        wsh[i] = src[i % (CH * QD)];
    }
    if (tid < 3 * CH) {
        const float* src = (tid < CH) ? bq : (tid < 2 * CH ? bk : bv);
        bsh[tid] = src[tid % CH];
    }
    for (int c = 0; c < QD; ++c)
        xs[c * 256 + tid] = x1[(size_t)c * L_TOT + pos0 + tid];
    __syncthreads();

    float xr[QD];
#pragma unroll
    for (int c = 0; c < QD; ++c) xr[c] = xs[c * 256 + tid];

    const size_t pos = (size_t)pos0 + tid;
    bf16 ov[CH];
    for (int g = 0; g < 3; ++g) {
        for (int o = 0; o < CH; ++o) {
            float a = bsh[g * CH + o];
#pragma unroll
            for (int c = 0; c < QD; ++c)
                a = fmaf(wsh[(g * CH + o) * QD + c], xr[c], a);
            ov[o] = (bf16)a;
        }
        bf16* dst = (g == 0) ? (qt + pos * CH)
                  : (g == 1) ? (kt + (pos + HALF) * CH)
                             : (vt + (pos + HALF) * CH);
        v8bf* d8 = (v8bf*)dst;
        const v8bf* s8 = (const v8bf*)ov;
        d8[0] = s8[0]; d8[1] = s8[1]; d8[2] = s8[2]; d8[3] = s8[3];
    }
}

// ------------------------------------------------ fused sliding attention
__global__ __launch_bounds__(256) void attn_kernel(
    const bf16* __restrict__ qt, const bf16* __restrict__ kt,
    const bf16* __restrict__ vt, const float* __restrict__ mask,
    const float* __restrict__ Wo, const float* __restrict__ bo,
    float* __restrict__ out)
{
    __shared__ bf16  k_lds[WW * KSTR];      // 80 KB  [key][ch], 16B pad/row
    __shared__ bf16  v_lds[CH * VSTR];      // ~64 KB [ch][key]
    __shared__ bf16  scrT[8][16 * SSTR];    // 10 KB  per-wave T = exp(s')
    __shared__ bf16  scrP[8][16 * SSTR];    // 10 KB  per-wave P = T * fmask
    __shared__ float bw_lds[WW];            // 4 KB   padded base mask
    __shared__ float lbw_lds[WW];           // 4 KB   log(base mask + 1e-9)

    const int tid   = threadIdx.x;
    const int lane  = tid & 31;
    const int wv    = tid >> 5;
    const int n     = blockIdx.x;           // sliding block
    const int qq    = blockIdx.y;           // query quarter (0..3)
    const int win0  = n * BLK;              // window start (padded coords)

    // ---- stage K window into LDS [key][ch] via the Tensor Data Mover:
    //      2D tile: 16 dwords/row x 1024 rows, LDS pad of 4 dwords per 16
    //      (pad_interval=3 -> 16 dwords, pad_amount=3 -> 4 dwords)  => 80B rows
#if __has_builtin(__builtin_amdgcn_tensor_load_to_lds)
    if (wv == 0) {
        const unsigned long long ga =
            (unsigned long long)(uintptr_t)(kt + (size_t)win0 * CH);
        const unsigned ldsa = (unsigned)(uintptr_t)(void*)&k_lds[0];
        u32x4 g0;
        g0[0] = 0x1u;                                   // count=1, load, user
        g0[1] = ldsa;                                   // lds_addr
        g0[2] = (unsigned)(ga & 0xFFFFFFFFu);           // global_addr[31:0]
        g0[3] = (unsigned)((ga >> 32) & 0x01FFFFFFu) | (2u << 30); // type=2
        i32x8 g1;
        g1[0] = (int)((2u << 16) | (1u << 20) | (3u << 22) | (3u << 25));
        g1[1] = (int)(16u << 16);                       // tensor_dim0=16 (lo16)
        g1[2] = (int)((16u >> 16) | ((unsigned)(LPAD & 0xFFFF) << 16));
        g1[3] = (int)(((unsigned)LPAD >> 16) | (16u << 16)); // tile_dim0=16
        g1[4] = (int)(1024u);                           // tile_dim1=1024, tile_dim2=0
        g1[5] = (int)(16u);                             // tensor_dim0_stride=16
        g1[6] = 0;
        g1[7] = 0;
        i32x4 gz; gz[0] = 0; gz[1] = 0; gz[2] = 0; gz[3] = 0;
#if defined(__clang_major__) && (__clang_major__ >= 23)
        i32x8 gz8;
#pragma unroll
        for (int i = 0; i < 8; ++i) gz8[i] = 0;
        __builtin_amdgcn_tensor_load_to_lds(g0, g1, gz, gz, gz8, 0);
#else
        __builtin_amdgcn_tensor_load_to_lds(g0, g1, gz, gz, 0);
#endif
        __builtin_amdgcn_s_wait_tensorcnt(0);
    }
#else
    for (int j = tid; j < WW; j += 256) {
        const v8bf* s8 = (const v8bf*)(kt + (size_t)(win0 + j) * CH);
        v8bf* d8 = (v8bf*)(k_lds + j * KSTR);
        d8[0] = s8[0]; d8[1] = s8[1]; d8[2] = s8[2]; d8[3] = s8[3];
    }
#endif
    // ---- stage V window transposed into LDS [ch][key]
    for (int j = tid; j < WW; j += 256) {
        const v8bf* s8 = (const v8bf*)(vt + (size_t)(win0 + j) * CH);
        bf16 tmp[CH];
        v8bf* t8 = (v8bf*)tmp;
        t8[0] = s8[0]; t8[1] = s8[1]; t8[2] = s8[2]; t8[3] = s8[3];
#pragma unroll
        for (int ch = 0; ch < CH; ++ch) v_lds[ch * VSTR + j] = tmp[ch];
    }
    // ---- stage padded base mask + its log for the window (log hoisted out
    //      of the hot loop: log(fm+1e-9) == wmask ? log(bw+1e-9) : log(1e-9))
    for (int j = tid; j < WW; j += 256) {
        const float bw = winmask(mask, win0 + j);
        bw_lds[j]  = bw;
        lbw_lds[j] = __logf(bw + 1e-9f);
    }
    __syncthreads();

    const int row   = lane & 15;            // A-frag row
    const int hi    = lane >> 4;
    const int kloA  = hi ? 16 : 0;          // A-layout K split
    const int kloB  = hi ? 16 : 0;          // B-layout K split
    const int colB  = lane & 15;            // B-layout N
    const int iblk0 = qq * 128 + wv * 16;   // query tile base within block
    const int qrow0 = n * BLK + iblk0;      // global query base
    const int Mrow  = 8 * hi;               // C-layout row base for this lane

    // ---- Q fragment (A: 16x32 bf16)
    const bf16* qb = qt + (size_t)(qrow0 + row) * CH;
    v16bf a_q = ld16(qb + kloA, qb + kloA + 8);

    // ones B-matrix: rowsum(T) via WMMA (replicated across all lanes of a row)
    v16bf b_ones;
#pragma unroll
    for (int i = 0; i < 16; ++i) b_ones[i] = (bf16)1.0f;

    v8f acc0, acc1, lsum;
#pragma unroll
    for (int r = 0; r < 8; ++r) { acc0[r] = 0.f; acc1[r] = 0.f; lsum[r] = 0.f; }

    const float scale = 0.17677669529663687f;   // 1/sqrt(32)
    bf16* myT = &scrT[wv][0];
    bf16* myP = &scrP[wv][0];

    for (int kc = 0; kc < 32; ++kc) {
        const int j0 = kc * 32 + colB;          // key col, tile 0
        const int j1 = j0 + 16;                 // key col, tile 1

        // energy: S = Q(16x32) x K(32x16), twice
        const bf16* kb0 = k_lds + j0 * KSTR + kloB;
        const bf16* kb1 = k_lds + j1 * KSTR + kloB;
        v16bf b_k0 = ld16(kb0, kb0 + 8);
        v16bf b_k1 = ld16(kb1, kb1 + 8);
        v8f S0, S1;
#pragma unroll
        for (int r = 0; r < 8; ++r) { S0[r] = 0.f; S1[r] = 0.f; }
        S0 = wmma_bf16(a_q, b_k0, S0);
        S1 = wmma_bf16(a_q, b_k1, S1);

        const float bw0  = bw_lds[j0];
        const float bw1  = bw_lds[j1];
        const float lbw0 = lbw_lds[j0];
        const float lbw1 = lbw_lds[j1];

        // T = exp(S*scale + log(fm+1e-9)); P = T * fm.  No running max:
        // energies are tightly bounded here, exp stays well inside fp32.
#pragma unroll
        for (int r = 0; r < 8; ++r) {
            const int lim = HALF + iblk0 + Mrow + r;        // causal limit
            const bool w0 = (j0 <= lim);
            const bool w1 = (j1 <= lim);
            const float t0 = __expf(S0[r] * scale + (w0 ? lbw0 : LOG_EPS));
            const float t1 = __expf(S1[r] * scale + (w1 ? lbw1 : LOG_EPS));
            myT[(Mrow + r) * SSTR + colB]      = (bf16)t0;
            myT[(Mrow + r) * SSTR + 16 + colB] = (bf16)t1;
            myP[(Mrow + r) * SSTR + colB]      = (bf16)(w0 ? t0 * bw0 : 0.f);
            myP[(Mrow + r) * SSTR + 16 + colB] = (bf16)(w1 ? t1 * bw1 : 0.f);
        }
        __builtin_amdgcn_wave_barrier();

        // rowsum accumulation: lsum += T(16x32) x ones(32x16)
        const bf16* tb = myT + row * SSTR;
        v16bf a_t = ld16(tb + kloA, tb + kloA + 8);
        lsum = wmma_bf16(a_t, b_ones, lsum);

        // O += P x V : B = V[32 keys][16 ch], twice (ch 0-15, 16-31)
        const bf16* pb = myP + row * SSTR;
        v16bf a_p = ld16(pb + kloA, pb + kloA + 8);
        const bf16* vb0 = v_lds + colB * VSTR + kc * 32 + kloB;
        const bf16* vb1 = v_lds + (16 + colB) * VSTR + kc * 32 + kloB;
        v16bf b_v0 = ld16(vb0, vb0 + 8);
        v16bf b_v1 = ld16(vb1, vb1 + 8);
        acc0 = wmma_bf16(a_p, b_v0, acc0);
        acc1 = wmma_bf16(a_p, b_v1, acc1);
        __builtin_amdgcn_wave_barrier();
    }

    // ---- normalize (fast rcp), ReLU, stash O (16 queries x 32 ch) as bf16
#pragma unroll
    for (int r = 0; r < 8; ++r) {
        const float inv = __builtin_amdgcn_rcpf(lsum[r]);
        myT[(Mrow + r) * SSTR + colB]      = (bf16)fmaxf(acc0[r] * inv, 0.f);
        myT[(Mrow + r) * SSTR + 16 + colB] = (bf16)fmaxf(acc1[r] * inv, 0.f);
    }
    __builtin_amdgcn_wave_barrier();

    // B = O^T (32 ch x 16 queries): lane N = query col, K contiguous ch
    const bf16* obse = myT + colB * SSTR + kloB;
    v16bf b_o = ld16(obse, obse + 8);

    const int  qg = qrow0 + colB;      // query column handled by this lane
    const float mk = mask[qg];

    // final: out = Wo(64x32) x O^T, 4 tiles of 16 output channels
    for (int og = 0; og < 4; ++og) {
        const float* wr = Wo + (size_t)(og * 16 + row) * CH;
        v16bf a_wo;
#pragma unroll
        for (int v = 0; v < 8; ++v) {
            a_wo[v]     = (bf16)wr[kloA + v];
            a_wo[v + 8] = (bf16)wr[kloA + 8 + v];
        }
        v8f F;
#pragma unroll
        for (int r = 0; r < 8; ++r) F[r] = 0.f;
        F = wmma_bf16(a_wo, b_o, F);
#pragma unroll
        for (int r = 0; r < 8; ++r) {
            const int o = og * 16 + r + 8 * hi;
            out[(size_t)o * L_TOT + qg] = (F[r] + bo[o]) * mk;
        }
    }
}

// ------------------------------------------------------------------- host
extern "C" void kernel_launch(void* const* d_in, const int* in_sizes, int n_in,
                              void* d_out, int out_size, void* d_ws, size_t ws_size,
                              hipStream_t stream) {
    (void)in_sizes; (void)n_in; (void)out_size; (void)ws_size;
    const float* x1 = (const float*)d_in[0];
    // d_in[1] = x2 (unused in encoder stage)
    const float* mask = (const float*)d_in[2];
    const float* Wq = (const float*)d_in[3];
    const float* bq = (const float*)d_in[4];
    const float* Wk = (const float*)d_in[5];
    const float* bk = (const float*)d_in[6];
    const float* Wv = (const float*)d_in[7];
    const float* bv = (const float*)d_in[8];
    const float* Wo = (const float*)d_in[9];
    const float* bo = (const float*)d_in[10];
    float* out = (float*)d_out;

    const size_t OFF_K = (size_t)L_TOT * CH * sizeof(bf16);         // 4 MB
    const size_t OFF_V = OFF_K + (size_t)LPAD * CH * sizeof(bf16);  // +~4 MB
    bf16* qt = (bf16*)d_ws;
    bf16* kt = (bf16*)((char*)d_ws + OFF_K);
    bf16* vt = (bf16*)((char*)d_ws + OFF_V);

    pad_kernel<<<128, 256, 0, stream>>>(kt, vt);
    proj_kernel<<<L_TOT / 256, 256, 0, stream>>>(x1, Wq, bq, Wk, bk, Wv, bv,
                                                 qt, kt, vt);
    attn_kernel<<<dim3(128, 4), 256, 0, stream>>>(qt, kt, vt, mask, Wo, bo, out);
}
